// TransformerBlock_1924145348849
// MI455X (gfx1250) — compile-verified, matrix operand order checked by reading, not run on hardware
//
#include <hip/hip_runtime.h>
#include <hip/hip_bf16.h>
#include <math.h>
#include <stdint.h>

// Problem constants (match reference)
#define BB   2
#define TT   2048
#define CC   1024
#define HH   16
#define DD   64
#define HID  4096
#define BT   (BB*TT)          // 4096 rows
#define EPSV 1e-5f

typedef _Float16 h16_t;
typedef __attribute__((ext_vector_type(8)))  _Float16 v8h;
typedef __attribute__((ext_vector_type(16))) _Float16 v16h;
typedef __attribute__((ext_vector_type(8)))  float    v8f;
typedef int vi4 __attribute__((vector_size(16)));
typedef __attribute__((address_space(1))) vi4* gvi4p;   // global int4*
typedef __attribute__((address_space(3))) vi4* lvi4p;   // LDS int4*

static __device__ __forceinline__ v16h mk16(v8h lo, v8h hi) {
    v16h r;
#pragma unroll
    for (int i = 0; i < 8; ++i) { r[i] = lo[i]; r[i + 8] = hi[i]; }
    return r;
}

// ---------------------------------------------------------------- async copy
#if __has_builtin(__builtin_amdgcn_global_load_async_to_lds_b128)
#define HAS_ASYNC_LDS 1
#else
#define HAS_ASYNC_LDS 0
#endif

// 16-byte global -> LDS copy. Async (ASYNCcnt) when the toolchain has the
// gfx1250 builtin; falls back to a synchronous reg round-trip otherwise.
// Flat address of __shared__ low 32 bits == LDS byte address (ISA 10.2
// aperture); integer casts build the as(1)/as(3)
// vector pointers the builtin expects.
static __device__ __forceinline__ void cp_b128(const h16_t* g, h16_t* l) {
#if HAS_ASYNC_LDS
    __builtin_amdgcn_global_load_async_to_lds_b128(
        (gvi4p)(uintptr_t)g, (lvi4p)(uint32_t)(uintptr_t)l, 0, 0);
#else
    *(v8h*)l = *(const v8h*)g;
#endif
}

template <int N>
static __device__ __forceinline__ void wait_cp() {
#if HAS_ASYNC_LDS
#if __has_builtin(__builtin_amdgcn_s_wait_asynccnt)
    __builtin_amdgcn_s_wait_asynccnt(N);
#else
    asm volatile("s_wait_asynccnt %0" :: "i"(N) : "memory");
#endif
#endif
}

// ---------------------------------------------------------------- weight cvt
// wt[n*K + k] = (f16) w[k*N + n]   (convert + transpose, once per launch)
__global__ __launch_bounds__(256) void cvt_w_t_kernel(
    const float* __restrict__ w, h16_t* __restrict__ wt,
    int K, int N, int lgK)
{
    const int total = K * N;
    for (int i = blockIdx.x * 256 + threadIdx.x; i < total; i += gridDim.x * 256) {
        const int k = i & (K - 1);
        const int n = i >> lgK;
        wt[(size_t)n * K + k] = (h16_t)w[(size_t)k * N + n];
    }
}

// ---------------------------------------------------------------- rmsnorm
__global__ __launch_bounds__(256) void rmsnorm_kernel(
    const float* __restrict__ x, const float* __restrict__ w,
    h16_t* __restrict__ out)
{
    __shared__ float red[8];
    __shared__ float stot;
    const int row = blockIdx.x;
    const float4 v = ((const float4*)(x + (size_t)row * CC))[threadIdx.x];
    float s = v.x * v.x + v.y * v.y + v.z * v.z + v.w * v.w;
#pragma unroll
    for (int off = 16; off; off >>= 1) s += __shfl_xor(s, off, 32);
    if ((threadIdx.x & 31) == 0) red[threadIdx.x >> 5] = s;
    __syncthreads();
    if (threadIdx.x == 0) {
        float t = 0.f;
#pragma unroll
        for (int i = 0; i < 8; ++i) t += red[i];
        stot = rsqrtf(t / (float)CC + EPSV);
    }
    __syncthreads();
    const float r = stot;
    const float4 wv = ((const float4*)w)[threadIdx.x];
    h16_t* op = out + (size_t)row * CC + threadIdx.x * 4;
    op[0] = (h16_t)(v.x * r * wv.x);
    op[1] = (h16_t)(v.y * r * wv.y);
    op[2] = (h16_t)(v.z * r * wv.z);
    op[3] = (h16_t)(v.w * r * wv.w);
}

// residual add + rmsnorm:  x1 = x + a;  m16 = rmsnorm(x1, w)
__global__ __launch_bounds__(256) void add_res_rmsnorm_kernel(
    const float* __restrict__ x, const float* __restrict__ a,
    const float* __restrict__ w, float* __restrict__ x1,
    h16_t* __restrict__ out)
{
    __shared__ float red[8];
    __shared__ float stot;
    const int row = blockIdx.x;
    const float4 xv = ((const float4*)(x + (size_t)row * CC))[threadIdx.x];
    const float4 av = ((const float4*)(a + (size_t)row * CC))[threadIdx.x];
    float4 t;
    t.x = xv.x + av.x; t.y = xv.y + av.y; t.z = xv.z + av.z; t.w = xv.w + av.w;
    ((float4*)(x1 + (size_t)row * CC))[threadIdx.x] = t;
    float s = t.x * t.x + t.y * t.y + t.z * t.z + t.w * t.w;
#pragma unroll
    for (int off = 16; off; off >>= 1) s += __shfl_xor(s, off, 32);
    if ((threadIdx.x & 31) == 0) red[threadIdx.x >> 5] = s;
    __syncthreads();
    if (threadIdx.x == 0) {
        float tt = 0.f;
#pragma unroll
        for (int i = 0; i < 8; ++i) tt += red[i];
        stot = rsqrtf(tt / (float)CC + EPSV);
    }
    __syncthreads();
    const float r = stot;
    const float4 wv = ((const float4*)w)[threadIdx.x];
    h16_t* op = out + (size_t)row * CC + threadIdx.x * 4;
    op[0] = (h16_t)(t.x * r * wv.x);
    op[1] = (h16_t)(t.y * r * wv.y);
    op[2] = (h16_t)(t.z * r * wv.z);
    op[3] = (h16_t)(t.w * r * wv.w);
}

__global__ __launch_bounds__(256) void final_add_kernel(
    const float* __restrict__ a, const float* __restrict__ b,
    float* __restrict__ out, int n)
{
    for (int i = blockIdx.x * 256 + threadIdx.x; i < n; i += gridDim.x * 256)
        out[i] = a[i] + b[i];
}

// ---------------------------------------------------------------- WMMA GEMM
// Out[M,N] = act(A[M,K] (f16, row-major) * Wt[N,K]^T (f16) + bias)
// Block: 256 thr (8 waves), tile BM=128 x BN=128, BK=64, double-buffered
// async LDS staging. Wave tile = 32x64 (2x4 WMMA fragments).
#define BM 128
#define BN 128
#define BK 64
#define LDT (BK + 8)

// stage a 128 x 64 f16 tile (rows from G, row stride ldg) into ls
static __device__ __forceinline__ void stage_tile(
    const h16_t* __restrict__ G, int ldg, h16_t* ls, int tid, int k0)
{
    const int row = tid >> 1;
    const int ch  = (tid & 1) * 32;
    const h16_t* gp = G + (size_t)row * ldg + k0 + ch;
    h16_t* lp = ls + row * LDT + ch;
#pragma unroll
    for (int j = 0; j < 4; ++j)
        cp_b128(gp + j * 8, lp + j * 8);
}

__global__ __launch_bounds__(256) void gemm_wmma_kernel(
    const h16_t* __restrict__ A, const h16_t* __restrict__ Wt,
    const float* __restrict__ bias,
    float* __restrict__ outF, int f32bhtd,
    h16_t* __restrict__ outH, h16_t* __restrict__ outHT,
    int M, int N, int K, int act)
{
    __shared__ h16_t lsA[2][BM * LDT];
    __shared__ h16_t lsB[2][BN * LDT];

    const int tid  = threadIdx.x;
    const int lane = tid & 31;
    const int wave = tid >> 5;
    const int m0 = blockIdx.y * BM;
    const int n0 = blockIdx.x * BN;
    const int wm = wave >> 1;         // 0..3
    const int wn = wave & 1;          // 0..1
    const int hl  = lane >> 4;
    const int l16 = lane & 15;

    const h16_t* Ab = A  + (size_t)m0 * K;
    const h16_t* Wb = Wt + (size_t)n0 * K;

    v8f acc[2][4];
#pragma unroll
    for (int i = 0; i < 2; ++i)
#pragma unroll
        for (int j = 0; j < 4; ++j)
            acc[i][j] = (v8f){0.f,0.f,0.f,0.f,0.f,0.f,0.f,0.f};

    const int nk = K / BK;
    stage_tile(Ab, K, lsA[0], tid, 0);
    stage_tile(Wb, K, lsB[0], tid, 0);

    for (int ki = 0; ki < nk; ++ki) {
        const int cur = ki & 1;
        if (ki + 1 < nk) {
            stage_tile(Ab, K, lsA[cur ^ 1], tid, (ki + 1) * BK);
            stage_tile(Wb, K, lsB[cur ^ 1], tid, (ki + 1) * BK);
            wait_cp<8>();             // current tile's 8 copies complete
        } else {
            wait_cp<0>();
        }
        __syncthreads();
#pragma unroll
        for (int kk = 0; kk < BK; kk += 32) {
            v16h afr[2], bfr[4];
#pragma unroll
            for (int mt = 0; mt < 2; ++mt) {
                const h16_t* p = &lsA[cur][(wm * 32 + mt * 16 + l16) * LDT + kk + hl * 8];
                afr[mt] = mk16(*(const v8h*)p, *(const v8h*)(p + 16));
            }
#pragma unroll
            for (int nt = 0; nt < 4; ++nt) {
                const h16_t* p = &lsB[cur][(wn * 64 + nt * 16 + l16) * LDT + kk + hl * 8];
                bfr[nt] = mk16(*(const v8h*)p, *(const v8h*)(p + 16));
            }
#pragma unroll
            for (int mt = 0; mt < 2; ++mt)
#pragma unroll
                for (int nt = 0; nt < 4; ++nt)
                    acc[mt][nt] = __builtin_amdgcn_wmma_f32_16x16x32_f16(
                        false, afr[mt], false, bfr[nt], (short)0, acc[mt][nt],
                        false, false);
        }
        __syncthreads();
    }

#pragma unroll
    for (int mt = 0; mt < 2; ++mt)
#pragma unroll
        for (int nt = 0; nt < 4; ++nt) {
            const int n = n0 + wn * 64 + nt * 16 + l16;
            const float bv = bias ? bias[n] : 0.f;
#pragma unroll
            for (int r = 0; r < 8; ++r) {
                const int m = m0 + wm * 32 + mt * 16 + r + hl * 8;
                float v = acc[mt][nt][r] + bv;
                if (act == 1) v = 0.5f * v * (1.f + erff(v * 0.70710678118f));
                if (outF) {
                    if (f32bhtd) {   // (BT,C) -> (B,H,T,D), only used when N==C
                        const int t = m & (TT - 1), bb = m >> 11;
                        const int hh = n >> 6, dd = n & (DD - 1);
                        outF[(((size_t)(bb * HH + hh) * TT) + t) * DD + dd] = v;
                    } else {
                        outF[(size_t)m * N + n] = v;
                    }
                }
                if (outH) outH[(size_t)m * N + n] = (h16_t)v;
                if (outHT) {         // per-head transposed f16: (B,H,D,T)
                    const int t = m & (TT - 1), bb = m >> 11;
                    const int hh = n >> 6, dd = n & (DD - 1);
                    outHT[((size_t)(bb * HH + hh) * DD + dd) * TT + t] = (h16_t)v;
                }
            }
        }
}

// ---------------------------------------------------------------- attention
// Block = 128 thr (4 waves) covering 64 queries of one (b,h); each wave owns a
// 16-query tile. K tile (32x64) and V^T tile (64x32) staged once per block,
// double-buffered via async-to-LDS. Flash-style online softmax per wave.
#define LKK 72   // lsK  row stride (64 + 8)
#define LVT 40   // lsVt row stride (32 + 8)
#define LPP 40   // lsP  row stride (32 + 8)

__global__ __launch_bounds__(128) void attn_kernel(
    const h16_t* __restrict__ Q, const h16_t* __restrict__ Kh,
    const h16_t* __restrict__ Vt, h16_t* __restrict__ Oout)
{
    __shared__ h16_t lsK[2][32 * LKK];   // [key][d]
    __shared__ h16_t lsVt[2][64 * LVT];  // [d][key]
    __shared__ h16_t lsP[4][16 * LPP];   // per-wave [q][key]

    const int b   = blockIdx.z;
    const int h   = blockIdx.y;
    const int q0b = blockIdx.x * 64;
    const int tid  = threadIdx.x;
    const int wave = tid >> 5;
    const int lane = tid & 31;
    const int hl  = lane >> 4;
    const int l16 = lane & 15;
    const int q0  = q0b + wave * 16;

    const h16_t* kbase = Kh + (size_t)(b * TT) * CC + h * DD;
    const h16_t* vbase = Vt + (size_t)(b * HH + h) * DD * TT;

    // Q fragments (A-matrix layout), D=64 -> 2 k-steps
    v16h qf[2];
    {
        const h16_t* qp = Q + ((size_t)(b * TT + q0 + l16)) * CC + h * DD;
#pragma unroll
        for (int kt = 0; kt < 2; ++kt) {
            const h16_t* p = qp + kt * 32 + hl * 8;
            qf[kt] = mk16(*(const v8h*)p, *(const v8h*)(p + 16));
        }
    }

    float mrow[8], lrow[8];
    v8f oacc[4];
#pragma unroll
    for (int r = 0; r < 8; ++r) { mrow[r] = -3.0e38f; lrow[r] = 0.f; }
#pragma unroll
    for (int dt = 0; dt < 4; ++dt)
        oacc[dt] = (v8f){0.f,0.f,0.f,0.f,0.f,0.f,0.f,0.f};

    const int nj = (q0b >> 5) + 2;       // key tiles of 32 up to q0b+63

    // stage K + V^T tile j0 into buffer `buf` (4 async b128 per thread)
    auto stage_kv = [&](int buf, int j0) {
        {
            const int row = tid >> 2, ch = (tid & 3) * 16;
            int krow = j0 + row; if (krow > TT - 1) krow = TT - 1;
            const h16_t* g = kbase + (size_t)krow * CC + ch;
            h16_t* l = &lsK[buf][row * LKK + ch];
            cp_b128(g, l); cp_b128(g + 8, l + 8);
        }
        {
            const int d = tid >> 1, ch = (tid & 1) * 16;
            int c0 = j0 + ch; if (c0 > TT - 16) c0 = TT - 16;
            const h16_t* g = vbase + (size_t)d * TT + c0;
            h16_t* l = &lsVt[buf][d * LVT + ch];
            cp_b128(g, l); cp_b128(g + 8, l + 8);
        }
    };

    stage_kv(0, 0);
    for (int jt = 0; jt < nj; ++jt) {
        const int cur = jt & 1;
        const int j0  = jt * 32;
        if (jt + 1 < nj) { stage_kv(cur ^ 1, (jt + 1) * 32); wait_cp<4>(); }
        else             { wait_cp<0>(); }
        __syncthreads();

        // S = Q * K^T (two 16-key n-tiles, two 32-d k-steps)
        v8f sacc[2];
#pragma unroll
        for (int nt = 0; nt < 2; ++nt) {
            sacc[nt] = (v8f){0.f,0.f,0.f,0.f,0.f,0.f,0.f,0.f};
#pragma unroll
            for (int kt = 0; kt < 2; ++kt) {
                const h16_t* p = &lsK[cur][(nt * 16 + l16) * LKK + kt * 32 + hl * 8];
                v16h bf = mk16(*(const v8h*)p, *(const v8h*)(p + 16));
                sacc[nt] = __builtin_amdgcn_wmma_f32_16x16x32_f16(
                    false, qf[kt], false, bf, (short)0, sacc[nt], false, false);
            }
        }

        // online softmax (row r+8*hl lives across the 16 lanes of a half)
        const int key0 = j0 + l16;
        const int key1 = j0 + 16 + l16;
        h16_t* pP = &lsP[wave][0];
#pragma unroll
        for (int r = 0; r < 8; ++r) {
            const int q = q0 + r + hl * 8;
            float s0 = sacc[0][r] * 0.125f;
            float s1 = sacc[1][r] * 0.125f;
            if (key0 > q) s0 = -1.0e30f;
            if (key1 > q) s1 = -1.0e30f;
            float vmax = fmaxf(s0, s1);
#pragma unroll
            for (int off = 1; off < 16; off <<= 1)
                vmax = fmaxf(vmax, __shfl_xor(vmax, off, 32));
            const float mnew = fmaxf(mrow[r], vmax);
            const float corr = __expf(mrow[r] - mnew);
            mrow[r] = mnew;
            const float p0 = __expf(s0 - mnew);
            const float p1 = __expf(s1 - mnew);
            float rs = p0 + p1;
#pragma unroll
            for (int off = 1; off < 16; off <<= 1)
                rs += __shfl_xor(rs, off, 32);
            lrow[r] = lrow[r] * corr + rs;
#pragma unroll
            for (int dt = 0; dt < 4; ++dt) oacc[dt][r] *= corr;
            pP[(r + hl * 8) * LPP + l16]      = (h16_t)p0;
            pP[(r + hl * 8) * LPP + 16 + l16] = (h16_t)p1;
        }

        // O += P * V (same-wave LDS round-trip; DS ops are in-order per wave)
        v16h pf;
        {
            const h16_t* p = &pP[l16 * LPP + hl * 8];
            pf = mk16(*(const v8h*)p, *(const v8h*)(p + 16));
        }
#pragma unroll
        for (int dt = 0; dt < 4; ++dt) {
            const h16_t* p = &lsVt[cur][(dt * 16 + l16) * LVT + hl * 8];
            v16h bf = mk16(*(const v8h*)p, *(const v8h*)(p + 16));
            oacc[dt] = __builtin_amdgcn_wmma_f32_16x16x32_f16(
                false, pf, false, bf, (short)0, oacc[dt], false, false);
        }
        __syncthreads();
    }

    // normalize + write (B,T,H*D) f16 for the O-projection GEMM
#pragma unroll
    for (int dt = 0; dt < 4; ++dt)
#pragma unroll
        for (int r = 0; r < 8; ++r) {
            const int q = q0 + r + hl * 8;
            const float val = oacc[dt][r] / lrow[r];
            Oout[((size_t)(b * TT + q)) * CC + h * DD + dt * 16 + l16] = (h16_t)val;
        }
}

// ---------------------------------------------------------------- launch
extern "C" void kernel_launch(void* const* d_in, const int* in_sizes, int n_in,
                              void* d_out, int out_size, void* d_ws, size_t ws_size,
                              hipStream_t stream) {
    const float* x   = (const float*)d_in[0];
    // d_in[1] = causal mask (implemented analytically)
    const float* anw = (const float*)d_in[2];
    const float* mnw = (const float*)d_in[3];
    const float* q_w = (const float*)d_in[4];
    const float* q_b = (const float*)d_in[5];
    const float* k_w = (const float*)d_in[6];
    const float* k_b = (const float*)d_in[7];
    const float* v_w = (const float*)d_in[8];
    const float* v_b = (const float*)d_in[9];
    const float* o_w = (const float*)d_in[10];
    const float* o_b = (const float*)d_in[11];
    const float* w1  = (const float*)d_in[12];
    const float* b1  = (const float*)d_in[13];
    const float* w2  = (const float*)d_in[14];
    const float* b2  = (const float*)d_in[15];

    float* outx = (float*)d_out;
    float* outk = outx + (size_t)BT * CC;
    float* outv = outk + (size_t)BT * CC;

    char* ws = (char*)d_ws;
    size_t off = 0;
    auto carve = [&](size_t bytes) -> void* {
        void* p = ws + off;
        off += (bytes + 255) & ~(size_t)255;
        return p;
    };
    h16_t* qwt  = (h16_t*)carve((size_t)CC * CC * 2);
    h16_t* kwt  = (h16_t*)carve((size_t)CC * CC * 2);
    h16_t* vwt  = (h16_t*)carve((size_t)CC * CC * 2);
    h16_t* owt  = (h16_t*)carve((size_t)CC * CC * 2);
    h16_t* w1t  = (h16_t*)carve((size_t)CC * HID * 2);
    h16_t* w2t  = (h16_t*)carve((size_t)HID * CC * 2);
    h16_t* h16p = (h16_t*)carve((size_t)BT * CC * 2);
    h16_t* q16p = (h16_t*)carve((size_t)BT * CC * 2);
    h16_t* k16p = (h16_t*)carve((size_t)BT * CC * 2);
    h16_t* v16t = (h16_t*)carve((size_t)BT * CC * 2);  // (B,H,D,T)
    h16_t* a16p = (h16_t*)carve((size_t)BT * CC * 2);
    float* aof  = (float*)carve((size_t)BT * CC * 4);
    float* x1   = (float*)carve((size_t)BT * CC * 4);
    h16_t* m16p = (h16_t*)carve((size_t)BT * CC * 2);
    h16_t* g16p = (h16_t*)carve((size_t)BT * HID * 2);
    float* mlpf = (float*)carve((size_t)BT * CC * 4);

    // 1. weight convert + transpose (Wt is N x K)
    cvt_w_t_kernel<<<1024, 256, 0, stream>>>(q_w, qwt, CC, CC, 10);
    cvt_w_t_kernel<<<1024, 256, 0, stream>>>(k_w, kwt, CC, CC, 10);
    cvt_w_t_kernel<<<1024, 256, 0, stream>>>(v_w, vwt, CC, CC, 10);
    cvt_w_t_kernel<<<1024, 256, 0, stream>>>(o_w, owt, CC, CC, 10);
    cvt_w_t_kernel<<<2048, 256, 0, stream>>>(w1, w1t, CC, HID, 10);
    cvt_w_t_kernel<<<2048, 256, 0, stream>>>(w2, w2t, HID, CC, 12);

    // 2. pre-attention rmsnorm
    rmsnorm_kernel<<<BT, 256, 0, stream>>>(x, anw, h16p);

    // 3. QKV projections (k/v fp32 outputs written directly in (B,H,T,D))
    dim3 gNC(CC / BN, BT / BM);
    gemm_wmma_kernel<<<gNC, 256, 0, stream>>>(h16p, qwt, q_b, nullptr, 0,
                                              q16p, nullptr, BT, CC, CC, 0);
    gemm_wmma_kernel<<<gNC, 256, 0, stream>>>(h16p, kwt, k_b, outk, 1,
                                              k16p, nullptr, BT, CC, CC, 0);
    gemm_wmma_kernel<<<gNC, 256, 0, stream>>>(h16p, vwt, v_b, outv, 1,
                                              nullptr, v16t, BT, CC, CC, 0);

    // 4. attention
    attn_kernel<<<dim3(TT / 64, HH, BB), 128, 0, stream>>>(q16p, k16p, v16t, a16p);

    // 5. output projection
    gemm_wmma_kernel<<<gNC, 256, 0, stream>>>(a16p, owt, o_b, aof, 0,
                                              nullptr, nullptr, BT, CC, CC, 0);

    // 6. residual + mlp rmsnorm
    add_res_rmsnorm_kernel<<<BT, 256, 0, stream>>>(x, aof, mnw, x1, m16p);

    // 7. MLP
    dim3 gNH(HID / BN, BT / BM);
    gemm_wmma_kernel<<<gNH, 256, 0, stream>>>(m16p, w1t, b1, nullptr, 0,
                                              g16p, nullptr, BT, HID, CC, 1);
    gemm_wmma_kernel<<<gNC, 256, 0, stream>>>(g16p, w2t, b2, mlpf, 0,
                                              nullptr, nullptr, BT, CC, HID, 0);

    // 8. final residual -> x output
    final_add_kernel<<<4096, 256, 0, stream>>>(x1, mlpf, outx, BT * CC);
}
